// RE_12146167513655
// MI455X (gfx1250) — matrix operand order repaired and checked by reading, not compiled, not run on hardware
//
#include <hip/hip_runtime.h>
#include <math.h>

#define NPTS 4096
#define NB   4
#define KNN  16

typedef __attribute__((ext_vector_type(16))) _Float16 v16h;
typedef __attribute__((ext_vector_type(8)))  float    v8f;

__device__ __forceinline__ void wait_lds() {
  asm volatile("s_wait_dscnt 0" ::: "memory");
}

// A-fragment (16x32 f16, M x K) from row-major half matrix W[ldk columns].
// Layout: lane&15 = M row; lanes0-15 hold K {0..15},
// lanes16-31 hold K {16..31}, packed 2 halves per VGPR.
__device__ __forceinline__ v16h load_a_frag(const _Float16* __restrict__ W,
                                            int ldk, int mrow, int kb, int lane) {
  const int M  = mrow + (lane & 15);
  const int kh = (lane >> 4) << 4;   // 0 or 16
  v16h a;
#pragma unroll
  for (int j = 0; j < 8; ++j) {
    int K = kb + kh + (j << 1);
    a[2*j]   = W[M * ldk + K];
    a[2*j+1] = W[M * ldk + K + 1];
  }
  return a;
}

// B-fragment (32x16 f16, K x N) from LDS stored [col][K] (K contiguous).
// lanes0-15: K=kb..kb+15 of column lane; lanes16-31: K=kb+16..kb+31.
__device__ __forceinline__ v16h load_b_frag(const _Float16* __restrict__ X,
                                            int ldk, int kb, int lane) {
  const int col = lane & 15;
  const int kh  = (lane >> 4) << 4;
  v16h b;
#pragma unroll
  for (int e = 0; e < 16; ++e) b[e] = X[col * ldk + kb + kh + e];
  return b;
}

__device__ __forceinline__ v8f wmma_f16(v16h a, v16h b, v8f c) {
  return __builtin_amdgcn_wmma_f32_16x16x32_f16(false, a, false, b,
                                                (short)0, c, false, false);
}

// ---------------- Kernel 1: fused pairwise-dist + top-16 kNN ----------------
__global__ __launch_bounds__(256) void knn_kernel(const float* __restrict__ xyz,
                                                  int* __restrict__ idx) {
  __shared__ float sx[256], sy[256], sz[256];
  const int tid = threadIdx.x;
  const int q = blockIdx.x * 256 + tid;          // 256 | N -> whole block same batch
  const int b = q / NPTS, n = q % NPTS;
  const float* xb = xyz + (size_t)b * 3 * NPTS;
  const float qx = xb[n], qy = xb[NPTS + n], qz = xb[2 * NPTS + n];
  float d[KNN]; int id[KNN];
#pragma unroll
  for (int i = 0; i < KNN; ++i) { d[i] = 3.4e38f; id[i] = 0; }
  for (int t = 0; t < NPTS; t += 256) {
    __syncthreads();
    sx[tid] = xb[t + tid];
    sy[tid] = xb[NPTS + t + tid];
    sz[tid] = xb[2 * NPTS + t + tid];
    if (t + 256 < NPTS) __builtin_prefetch(&xb[t + 256 + tid], 0, 0);
    __syncthreads();
    for (int j = 0; j < 256; ++j) {
      float dx = qx - sx[j], dy = qy - sy[j], dz = qz - sz[j];
      float dd = dx * dx + dy * dy + dz * dz;
      if (dd < d[KNN - 1]) {            // sorted insert; strict < keeps lower index on ties
        float cd = dd; int ci = t + j;
#pragma unroll
        for (int i = 0; i < KNN; ++i) {
          bool c = cd < d[i];
          float td = d[i]; int ti = id[i];
          d[i]  = c ? cd : td;  id[i] = c ? ci : ti;
          cd    = c ? td : cd;  ci    = c ? ti : ci;
        }
      }
    }
  }
#pragma unroll
  for (int i = 0; i < KNN; ++i) idx[(size_t)q * KNN + i] = id[i];
}

// ---------------- Kernel 2a: LocalShape "planes" (max_k t*|t|/nrm) ----------
__global__ __launch_bounds__(64) void planes_kernel(const float* __restrict__ xyz,
                                                    const int* __restrict__ idx,
                                                    const float* __restrict__ wp,
                                                    float* __restrict__ planes) {
  __shared__ float rel[3][15];
  __shared__ float nrm[15];
  const int p = blockIdx.x, b = p / NPTS, n = p % NPTS;
  const int tid = threadIdx.x;
  const float* xb = xyz + (size_t)b * 3 * NPTS;
  if (tid < 15) {
    int j = idx[(size_t)p * KNN + 1 + tid];       // exclude self
    float dx = xb[j] - xb[n];
    float dy = xb[NPTS + j] - xb[NPTS + n];
    float dz = xb[2 * NPTS + j] - xb[2 * NPTS + n];
    rel[0][tid] = dx; rel[1][tid] = dy; rel[2][tid] = dz;
    nrm[tid] = sqrtf(dx * dx + dy * dy + dz * dz) + 1e-8f;
  }
  __syncthreads();
  float w0 = wp[tid * 3], w1 = wp[tid * 3 + 1], w2 = wp[tid * 3 + 2];
  float m = -3.4e38f;
#pragma unroll
  for (int k = 0; k < 15; ++k) {
    float t = w0 * rel[0][k] + w1 * rel[1][k] + w2 * rel[2][k];
    m = fmaxf(m, t * fabsf(t) / nrm[k]);          // nrm*(t/nrm)*|t/nrm|
  }
  planes[((size_t)b * 64 + tid) * NPTS + n] = m;
}

// ---------------- Kernel 3: mlp1 3->32->64 (per point) ----------------------
__global__ __launch_bounds__(256) void mlp1_kernel(const float* __restrict__ xyz,
    const float* __restrict__ w1a, const float* __restrict__ g1a, const float* __restrict__ b1a,
    const float* __restrict__ w1b, const float* __restrict__ g1b, const float* __restrict__ b1b,
    float* __restrict__ f) {
  __shared__ float sw1a[96], sg1a[32], sb1a[32];
  __shared__ float sw1b[2048], sgb[64], sbb[64];
  const int tid = threadIdx.x;
  for (int i = tid; i < 96; i += 256) sw1a[i] = w1a[i];
  if (tid < 32) { sg1a[tid] = g1a[tid]; sb1a[tid] = b1a[tid]; }
  for (int i = tid; i < 2048; i += 256) sw1b[i] = w1b[i];
  if (tid < 64) { sgb[tid] = g1b[tid]; sbb[tid] = b1b[tid]; }
  __syncthreads();
  const int p = blockIdx.x * 256 + tid, b = p / NPTS, n = p % NPTS;
  const float* xb = xyz + (size_t)b * 3 * NPTS;
  float x0 = xb[n], x1 = xb[NPTS + n], x2 = xb[2 * NPTS + n];
  float h[32];
#pragma unroll
  for (int o = 0; o < 32; ++o) {
    float a = sw1a[o * 3] * x0 + sw1a[o * 3 + 1] * x1 + sw1a[o * 3 + 2] * x2;
    h[o] = fmaxf(a * sg1a[o] + sb1a[o], 0.f);
  }
  for (int o = 0; o < 64; ++o) {
    float a = 0.f;
#pragma unroll
    for (int i = 0; i < 32; ++i) a += sw1b[o * 32 + i] * h[i];
    f[((size_t)b * 64 + o) * NPTS + n] = fmaxf(a * sgb[o] + sbb[o], 0.f);
  }
}

// ---------------- Kernel 2b: f_shapes = relu(bn(ws @ planes)) via WMMA ------
__global__ __launch_bounds__(128) void fshapes_kernel(const float* __restrict__ planes,
    const float* __restrict__ wsw, const float* __restrict__ gs, const float* __restrict__ bs,
    float* __restrict__ fcat) {
  __shared__ _Float16 wLds[64 * 64];
  __shared__ float sbs[64];
  const int tid = threadIdx.x;
  for (int i = tid; i < 64 * 64; i += 128) wLds[i] = (_Float16)(wsw[i] * gs[i >> 6]);
  if (tid < 64) sbs[tid] = bs[tid];
  __syncthreads();
  const int wave = tid >> 5, lane = tid & 31;
  const int colbase = (blockIdx.x * 4 + wave) * 16;
  const int col = colbase + (lane & 15);
  const int b = col / NPTS, n = col % NPTS;
  const float* pb = planes + (size_t)b * 64 * NPTS;
  v16h bf[2];
#pragma unroll
  for (int t = 0; t < 2; ++t) {
    int kb = t * 32 + ((lane >> 4) << 4);
#pragma unroll
    for (int e = 0; e < 16; ++e) bf[t][e] = (_Float16)pb[(size_t)(kb + e) * NPTS + n];
  }
#pragma unroll
  for (int mt = 0; mt < 4; ++mt) {
    v8f c = {};
    c = wmma_f16(load_a_frag(wLds, 64, mt * 16, 0, lane), bf[0], c);
    c = wmma_f16(load_a_frag(wLds, 64, mt * 16, 32, lane), bf[1], c);
#pragma unroll
    for (int r = 0; r < 8; ++r) {
      int M = mt * 16 + r + ((lane >> 4) << 3);
      fcat[((size_t)b * 128 + 64 + M) * NPTS + n] = fmaxf(c[r] + sbs[M], 0.f);
    }
  }
}

// ---------------- Kernel 4: neighbor attention (WMMA GEMMs + softmax) -------
__global__ __launch_bounds__(128) void attn_kernel(const float* __restrict__ f,
    const int* __restrict__ idx,
    const float* __restrict__ wa1, const float* __restrict__ ga1, const float* __restrict__ ba1,
    const float* __restrict__ wa2, const float* __restrict__ ga2, const float* __restrict__ ba2,
    float* __restrict__ fcat) {
  __shared__ _Float16 w1Lds[128 * 64];
  __shared__ _Float16 w2Lds[64 * 128];
  __shared__ float sba1[128], sba2[64];
  __shared__ float    fknn[4][16 * 64];   // per-wave f32 f_knn [k][c]
  __shared__ _Float16 hLds[4][16 * 128];  // per-wave h [col][K]
  const int tid = threadIdx.x;
  for (int i = tid; i < 128 * 64; i += 128) w1Lds[i] = (_Float16)(wa1[i] * ga1[i >> 6]);
  for (int i = tid; i < 64 * 128; i += 128) w2Lds[i] = (_Float16)(wa2[i] * ga2[i >> 7]);
  if (tid < 128) sba1[tid] = ba1[tid];
  if (tid < 64)  sba2[tid] = ba2[tid];
  __syncthreads();
  const int wave = tid >> 5, lane = tid & 31;
  const int p = blockIdx.x * 4 + wave;    // one wave = one point
  const int b = p / NPTS, n = p % NPTS;
  const float* fb = f + (size_t)b * 64 * NPTS;
  const int k = lane & 15;
  const int nb = idx[(size_t)p * KNN + k];
  float*    fkw = fknn[wave];
  _Float16* hw  = hLds[wave];
  // Build f_knn = f[:,idx]-f[:,n] directly in B-fragment layout (+ f32 copy in LDS)
  v16h bf[2];
  const int kh = (lane >> 4) << 4;
#pragma unroll
  for (int t = 0; t < 2; ++t) {
#pragma unroll
    for (int e = 0; e < 16; ++e) {
      int c = t * 32 + kh + e;
      float v = fb[(size_t)c * NPTS + nb] - fb[(size_t)c * NPTS + n];
      bf[t][e] = (_Float16)v;
      fkw[k * 64 + c] = v;
    }
  }
  // GEMM1: h(128x16) = relu(ga1*wa1 @ f_knn + ba1)
#pragma unroll
  for (int mt = 0; mt < 8; ++mt) {
    v8f c8 = {};
    c8 = wmma_f16(load_a_frag(w1Lds, 64, mt * 16, 0, lane), bf[0], c8);
    c8 = wmma_f16(load_a_frag(w1Lds, 64, mt * 16, 32, lane), bf[1], c8);
#pragma unroll
    for (int r = 0; r < 8; ++r) {
      int M = mt * 16 + r + ((lane >> 4) << 3);
      hw[k * 128 + M] = (_Float16)fmaxf(c8[r] + sba1[M], 0.f);
    }
  }
  wait_lds();   // wave-private LDS: order cross-lane store->load
  // GEMM2: logits(64x16) = ga2*wa2 @ h + ba2 ; softmax over k; weighted sum
#pragma unroll
  for (int mt = 0; mt < 4; ++mt) {
    v8f acc = {};
#pragma unroll
    for (int ks = 0; ks < 4; ++ks)
      acc = wmma_f16(load_a_frag(w2Lds, 128, mt * 16, ks * 32, lane),
                     load_b_frag(hw, 128, ks * 32, lane), acc);
#pragma unroll
    for (int r = 0; r < 8; ++r) {
      int M = mt * 16 + r + ((lane >> 4) << 3);
      float l = acc[r] + sba2[M];
      float m = l;
#pragma unroll
      for (int s = 8; s >= 1; s >>= 1) m = fmaxf(m, __shfl_xor(m, s, 16));
      float e = __expf(l - m);
      float fk = fkw[k * 64 + M];
      float num = e * fk, den = e;
#pragma unroll
      for (int s = 8; s >= 1; s >>= 1) {
        num += __shfl_xor(num, s, 16);
        den += __shfl_xor(den, s, 16);
      }
      if (k == 0) fcat[((size_t)b * 128 + M) * NPTS + n] = num / den;
    }
  }
}

// ---------------- Kernel 5: mlp3 128->128->6 + output assembly --------------
__global__ __launch_bounds__(128) void mlp3_kernel(const float* __restrict__ fcat,
    const float* __restrict__ xyz,
    const float* __restrict__ w3a, const float* __restrict__ g3a, const float* __restrict__ b3a,
    const float* __restrict__ w3b, const float* __restrict__ g3b, const float* __restrict__ b3b,
    float* __restrict__ out) {
  __shared__ _Float16 wLds[128 * 128];
  __shared__ float sb3a[128];
  __shared__ _Float16 hLds[4][16 * 128];
  __shared__ float sw3b[6 * 128], sg3b[6], sb3b[6];
  const int tid = threadIdx.x;
  for (int i = tid; i < 128 * 128; i += 128) wLds[i] = (_Float16)(w3a[i] * g3a[i >> 7]);
  if (tid < 128) sb3a[tid] = b3a[tid];
  for (int i = tid; i < 6 * 128; i += 128) sw3b[i] = w3b[i];
  if (tid < 6) { sg3b[tid] = g3b[tid]; sb3b[tid] = b3b[tid]; }
  __syncthreads();
  const int wave = tid >> 5, lane = tid & 31;
  const int colbase = (blockIdx.x * 4 + wave) * 16;
  const int col = colbase + (lane & 15);
  const int b = col / NPTS, n = col % NPTS;
  const float* xb = fcat + (size_t)b * 128 * NPTS;
  v16h bf[4];
#pragma unroll
  for (int t = 0; t < 4; ++t) {
    int kb = t * 32 + ((lane >> 4) << 4);
#pragma unroll
    for (int e = 0; e < 16; ++e) bf[t][e] = (_Float16)xb[(size_t)(kb + e) * NPTS + n];
  }
  _Float16* hw = hLds[wave];
#pragma unroll
  for (int mt = 0; mt < 8; ++mt) {
    v8f c8 = {};
#pragma unroll
    for (int ks = 0; ks < 4; ++ks)
      c8 = wmma_f16(load_a_frag(wLds, 128, mt * 16, ks * 32, lane), bf[ks], c8);
#pragma unroll
    for (int r = 0; r < 8; ++r) {
      int M = mt * 16 + r + ((lane >> 4) << 3);
      hw[(lane & 15) * 128 + M] = (_Float16)fmaxf(c8[r] + sb3a[M], 0.f);
    }
  }
  wait_lds();
  // tail: 6x128 conv + bn, then out[b, c>>1, (c&1)*N + n] = xyz + 0.15*val
  for (int t = lane; t < 96; t += 32) {
    int o = t >> 4, cc = t & 15;
    int p2 = colbase + cc, b2 = p2 / NPTS, n2 = p2 % NPTS;
    float acc = 0.f;
#pragma unroll
    for (int K = 0; K < 128; ++K) acc += sw3b[o * 128 + K] * (float)hw[cc * 128 + K];
    float v = sg3b[o] * acc + sb3b[o];
    int cp = o >> 1, m = ((o & 1) << 12) + n2;
    out[((size_t)b2 * 3 + cp) * (2 * NPTS) + m] =
        xyz[((size_t)b2 * 3 + cp) * NPTS + n2] + 0.15f * v;
  }
}

extern "C" void kernel_launch(void* const* d_in, const int* in_sizes, int n_in,
                              void* d_out, int out_size, void* d_ws, size_t ws_size,
                              hipStream_t stream) {
  (void)in_sizes; (void)n_in; (void)out_size; (void)ws_size;
  const float* xyz = (const float*)d_in[0];
  const float* wp  = (const float*)d_in[1];
  const float* wsw = (const float*)d_in[2];
  const float* gs  = (const float*)d_in[3];
  const float* bs  = (const float*)d_in[4];
  const float* w1a = (const float*)d_in[5];
  const float* g1a = (const float*)d_in[6];
  const float* b1a = (const float*)d_in[7];
  const float* w1b = (const float*)d_in[8];
  const float* g1b = (const float*)d_in[9];
  const float* b1b = (const float*)d_in[10];
  const float* wa1 = (const float*)d_in[11];
  const float* ga1 = (const float*)d_in[12];
  const float* ba1 = (const float*)d_in[13];
  const float* wa2 = (const float*)d_in[14];
  const float* ga2 = (const float*)d_in[15];
  const float* ba2 = (const float*)d_in[16];
  const float* w3a = (const float*)d_in[17];
  const float* g3a = (const float*)d_in[18];
  const float* b3a = (const float*)d_in[19];
  const float* w3b = (const float*)d_in[20];
  const float* g3b = (const float*)d_in[21];
  const float* b3b = (const float*)d_in[22];

  char* ws = (char*)d_ws;
  int*   idx    = (int*)ws;                                   // 16384*16 int  = 1 MB
  float* f      = (float*)(ws + (1u << 20));                  // B*64*N f32    = 4 MB
  float* planes = (float*)(ws + (1u << 20) + (4u << 20));     // B*64*N f32    = 4 MB
  float* fcat   = (float*)(ws + (1u << 20) + (8u << 20));     // B*128*N f32   = 8 MB
  float* outp   = (float*)d_out;

  knn_kernel   <<<NB * NPTS / 256, 256, 0, stream>>>(xyz, idx);
  planes_kernel<<<NB * NPTS,        64, 0, stream>>>(xyz, idx, wp, planes);
  mlp1_kernel  <<<NB * NPTS / 256, 256, 0, stream>>>(xyz, w1a, g1a, b1a, w1b, g1b, b1b, f);
  fshapes_kernel<<<NB * NPTS / 64, 128, 0, stream>>>(planes, wsw, gs, bs, fcat);
  attn_kernel  <<<NB * NPTS / 4,   128, 0, stream>>>(f, idx, wa1, ga1, ba1, wa2, ga2, ba2, fcat);
  mlp3_kernel  <<<NB * NPTS / 64,  128, 0, stream>>>(fcat, xyz, w3a, g3a, b3a, w3b, g3b, b3b, outp);
}
